// MXFP8GroupedExperts_37014028157134
// MI455X (gfx1250) — compile-verified, matrix operand order checked by reading, not run on hardware
//
#include <hip/hip_runtime.h>

// MXFP8 grouped experts MLP for MI455X (gfx1250, wave32).
// GEMM1 (x @ w13^T, MX scale fused, SiLU gate fused) -> bf16 ws   [WMMA bf16]
// GEMM2 (h @ w2^T) -> fp32 out                                    [WMMA bf16 + TDM]
// GEMM2's A tile (bf16 activations) is staged into LDS by the Tensor Data
// Mover (tensor_load_to_lds) with hardware row padding matching the padded
// LDS layout; completion is ordered with s_wait_tensorcnt.

typedef __attribute__((ext_vector_type(16))) __bf16 v16bf;
typedef __attribute__((ext_vector_type(8)))  __bf16 bf16x8;
typedef __attribute__((ext_vector_type(8)))  float  v8f;
typedef unsigned int u32;
typedef u32 u32x4 __attribute__((ext_vector_type(4)));
typedef u32 u32x8 __attribute__((ext_vector_type(8)));

#define D_DIM 1024
#define H_DIM 512
#define N_EXP 8
#define TPE   1024
#define SBLK  32

#define BM 128
#define BN 64
#define BK 32
#define LDA (BK + 8)   // padded LDS row: 80B stride, 16B aligned, conflict-free

union Frag { v16bf v; bf16x8 h[2]; };

__device__ __forceinline__ float silu_f(float x) {
    return x / (1.0f + __expf(-x));
}

__device__ __forceinline__ v8f vzero8() {
    v8f z;
#pragma unroll
    for (int i = 0; i < 8; ++i) z[i] = 0.0f;
    return z;
}

// ---------------------------------------------------------------------------
// TDM: issue one tensor_load_to_lds for a BM x BK bf16 tile.
// D# group1/2/3 are loop-invariant; group0 carries lds/global addresses.
// ---------------------------------------------------------------------------
__device__ __forceinline__ void tdm_load_tile(u32 lds_addr, unsigned long long gaddr,
                                              u32x8 g1, u32x4 gz) {
    u32x4 g0;
    g0[0] = 1u;                                   // count=1, user mode, no gather
    g0[1] = lds_addr;                             // lds_addr [63:32]
    g0[2] = (u32)gaddr;                           // global_addr [95:64]
    g0[3] = (u32)((gaddr >> 32) & 0x01FFFFFFu)    // global_addr [120:96]
          | 0x80000000u;                          // type=2 ("image") [127:126]
    asm volatile("tensor_load_to_lds %0, %1, %2, %3"
                 :: "s"(g0), "s"(g1), "s"(gz), "s"(gz)
                 : "memory");
}

// ---------------------------------------------------------------------------
// GEMM1 + SiLU gate: per expert e, h[t,n] = silu(x@w1^T) * (x@w3^T)
// x rows get the MX e8m0 scale applied during the LDS staging convert.
// ---------------------------------------------------------------------------
__global__ __launch_bounds__(256)
void mx_gemm1_silu(const float* __restrict__ x,
                   const float* __restrict__ scales,
                   const float* __restrict__ w13,
                   __bf16* __restrict__ hws) {
    __shared__ __bf16 sA[BM][LDA];
    __shared__ __bf16 sB1[BN][LDA];
    __shared__ __bf16 sB3[BN][LDA];

    const int tid  = threadIdx.x;
    const int e    = blockIdx.z;
    const int bm0  = blockIdx.x * BM;
    const int bn0  = blockIdx.y * BN;

    const int lane = tid & 31;
    const int wv   = tid >> 5;          // wave id 0..7 -> 16-row strip
    const int half = lane >> 4;         // lane group selects K interleave
    const int l15  = lane & 15;
    const int kb   = half * 8;          // K base: 0 or 8 (then +16 for hi half)

    // cooperative load mapping: A tile 128x32 f32, B tiles 2x(64x32) f32
    const int arow = tid >> 1;
    const int ac0  = (tid & 1) * 16;
    const int bsel = tid >> 7;          // 0 -> w1 rows, 1 -> w3 rows
    const int bj   = tid & 127;
    const int brow = bj >> 1;
    const int bc0  = (bj & 1) * 16;

    const size_t tokA = (size_t)(e * TPE + bm0 + arow);
    const float* __restrict__ xrow = x + tokA * D_DIM + ac0;
    const float* __restrict__ srow = scales + tokA * (D_DIM / SBLK);
    const int    wrow = (bsel ? H_DIM : 0) + bn0 + brow;
    const float* __restrict__ wrp  = w13 + ((size_t)e * 2 * H_DIM + wrow) * D_DIM + bc0;
    __bf16 (*sB)[LDA] = bsel ? sB3 : sB1;

    v8f acc1[4], acc3[4];
#pragma unroll
    for (int i = 0; i < 4; ++i) { acc1[i] = vzero8(); acc3[i] = vzero8(); }

    for (int k0 = 0; k0 < D_DIM; k0 += BK) {
        // ---- stage tiles: global f32 -> (scale) -> bf16 LDS ----
        {
            const float s = srow[k0 >> 5];         // one MX scale per 32-wide k-chunk
            const float4* xp = (const float4*)(xrow + k0);
            float fb[16];
            *(float4*)&fb[0]  = xp[0];
            *(float4*)&fb[4]  = xp[1];
            *(float4*)&fb[8]  = xp[2];
            *(float4*)&fb[12] = xp[3];
#pragma unroll
            for (int j = 0; j < 16; ++j) sA[arow][ac0 + j] = (__bf16)(fb[j] * s);

            const float4* wp = (const float4*)(wrp + k0);
            *(float4*)&fb[0]  = wp[0];
            *(float4*)&fb[4]  = wp[1];
            *(float4*)&fb[8]  = wp[2];
            *(float4*)&fb[12] = wp[3];
#pragma unroll
            for (int j = 0; j < 16; ++j) sB[brow][bc0 + j] = (__bf16)fb[j];

            if (k0 + BK < D_DIM) {
                __builtin_prefetch(xrow + k0 + BK, 0, 0);   // global_prefetch_b8
                __builtin_prefetch(wrp  + k0 + BK, 0, 0);
            }
        }
        __syncthreads();

        // ---- WMMA: 1 A-fragment reused across 8 B-fragments ----
        // 16-bit A layout: lane<16 holds K 0..7 & 16..23; lane>=16 K 8..15 & 24..31
        Frag a;
        a.h[0] = *(const bf16x8*)&sA[wv * 16 + l15][kb];
        a.h[1] = *(const bf16x8*)&sA[wv * 16 + l15][kb + 16];
#pragma unroll
        for (int nt = 0; nt < 4; ++nt) {
            Frag b;
            b.h[0] = *(const bf16x8*)&sB1[nt * 16 + l15][kb];
            b.h[1] = *(const bf16x8*)&sB1[nt * 16 + l15][kb + 16];
            acc1[nt] = __builtin_amdgcn_wmma_f32_16x16x32_bf16(
                false, a.v, false, b.v, (short)0, acc1[nt], false, false);
            b.h[0] = *(const bf16x8*)&sB3[nt * 16 + l15][kb];
            b.h[1] = *(const bf16x8*)&sB3[nt * 16 + l15][kb + 16];
            acc3[nt] = __builtin_amdgcn_wmma_f32_16x16x32_bf16(
                false, a.v, false, b.v, (short)0, acc3[nt], false, false);
        }
        __syncthreads();
    }

    // ---- fused SiLU gate, store bf16 activations to workspace ----
    // C layout: VGPR r -> row r (lanes 0-15) / row r+8 (lanes 16-31), col = l15
    const int mrow = bm0 + wv * 16 + half * 8;
#pragma unroll
    for (int nt = 0; nt < 4; ++nt) {
        const int ncol = bn0 + nt * 16 + l15;
#pragma unroll
        for (int r = 0; r < 8; ++r) {
            const float g = silu_f(acc1[nt][r]) * acc3[nt][r];
            hws[((size_t)e * TPE + mrow + r) * H_DIM + ncol] = (__bf16)g;
        }
    }
}

// ---------------------------------------------------------------------------
// GEMM2: out[t,d] = sum_h hws[t,h] * w2[e][d,h]   (fp32 out)
// A (bf16 activations) staged by TDM; B (w2 f32->bf16) staged cooperatively.
// ---------------------------------------------------------------------------
__global__ __launch_bounds__(256)
void mx_gemm2(const __bf16* __restrict__ hws,
              const float* __restrict__ w2,
              float* __restrict__ out) {
    __shared__ __bf16 sA[BM][LDA];
    __shared__ __bf16 sB[BN][LDA];

    const int tid  = threadIdx.x;
    const int e    = blockIdx.z;
    const int bm0  = blockIdx.x * BM;
    const int bn0  = blockIdx.y * BN;

    const int lane = tid & 31;
    const int wv   = tid >> 5;
    const int half = lane >> 4;
    const int l15  = lane & 15;
    const int kb   = half * 8;

    const int brow = tid >> 2;
    const int bc0  = (tid & 3) * 8;

    const float* __restrict__ wrp =
        w2 + ((size_t)e * D_DIM + bn0 + brow) * H_DIM + bc0;

    // ---- loop-invariant TDM descriptor pieces (wave-uniform) ----
    // group1: data_size=2B, pad_enable, pad_interval=3 (16 DWORD rows),
    //         pad_amount=3 (4 DWORDs = 16B) -> LDS row stride 80B == LDA.
    u32x8 g1;
    g1[0] = (1u << 16) | (1u << 20) | (3u << 22) | (3u << 25);
    g1[1] = (u32)(H_DIM & 0xFFFF) << 16;          // tensor_dim0 = 512  [79:48]
    g1[2] = (u32)(TPE   & 0xFFFF) << 16;          // tensor_dim1 = 1024 [111:80]
    g1[3] = (u32)BK << 16;                        // tile_dim0 = 32     [127:112]
    g1[4] = (u32)BM;                              // tile_dim1 = 128    [143:128]
    g1[5] = (u32)H_DIM;                           // tensor_dim0_stride [207:160]
    g1[6] = 0u;
    g1[7] = 0u;                                   // tensor_dim1_stride unused (2D)
    u32x4 gz;
    gz[0] = 0u; gz[1] = 0u; gz[2] = 0u; gz[3] = 0u;   // groups 2/3: dims unused

    const u32 ldsA = (u32)(unsigned long long)(uintptr_t)&sA[0][0];
    const unsigned long long gA =
        (unsigned long long)(uintptr_t)(hws + ((size_t)e * TPE + bm0) * H_DIM);

    v8f acc[4];
#pragma unroll
    for (int i = 0; i < 4; ++i) acc[i] = vzero8();

    for (int k0 = 0; k0 < H_DIM; k0 += BK) {
        // ---- TDM: DMA the 128x32 bf16 A tile into padded LDS (wave 0) ----
        if (wv == 0) {
            tdm_load_tile(ldsA, gA + (unsigned long long)k0 * 2u, g1, gz);
        }
        // ---- cooperative B staging overlaps the DMA: w2 f32 -> bf16 ----
        {
            const float4* wp = (const float4*)(wrp + k0);
            float fb[8];
            *(float4*)&fb[0] = wp[0];
            *(float4*)&fb[4] = wp[1];
#pragma unroll
            for (int j = 0; j < 8; ++j) sB[brow][bc0 + j] = (__bf16)fb[j];

            if (k0 + BK < H_DIM) __builtin_prefetch(wrp + k0 + BK, 0, 0);
        }
        if (wv == 0) __builtin_amdgcn_s_wait_tensorcnt(0);   // A tile landed
        __syncthreads();

        Frag a;
        a.h[0] = *(const bf16x8*)&sA[wv * 16 + l15][kb];
        a.h[1] = *(const bf16x8*)&sA[wv * 16 + l15][kb + 16];
#pragma unroll
        for (int nt = 0; nt < 4; ++nt) {
            Frag b;
            b.h[0] = *(const bf16x8*)&sB[nt * 16 + l15][kb];
            b.h[1] = *(const bf16x8*)&sB[nt * 16 + l15][kb + 16];
            acc[nt] = __builtin_amdgcn_wmma_f32_16x16x32_bf16(
                false, a.v, false, b.v, (short)0, acc[nt], false, false);
        }
        __syncthreads();   // nobody still reads sA when next TDM is issued
    }

    const int mrow = bm0 + wv * 16 + half * 8;
#pragma unroll
    for (int nt = 0; nt < 4; ++nt) {
        const int ncol = bn0 + nt * 16 + l15;
#pragma unroll
        for (int r = 0; r < 8; ++r) {
            out[((size_t)e * TPE + mrow + r) * D_DIM + ncol] = acc[nt][r];
        }
    }
}

// ---------------------------------------------------------------------------
extern "C" void kernel_launch(void* const* d_in, const int* in_sizes, int n_in,
                              void* d_out, int out_size, void* d_ws, size_t ws_size,
                              hipStream_t stream) {
    const float* x      = (const float*)d_in[0];   // (T, D) fp32
    const float* scales = (const float*)d_in[1];   // (T, D/32) fp32 (e8m0 values)
    // d_in[2] num_tokens_per_expert, d_in[3] expert_padded_offsets:
    // reference semantics are a plain reshape (T = E * TPE contiguous), so the
    // uniform layout is baked into the grid.
    const float* w13    = (const float*)d_in[4];   // (E, 2H, D) fp32
    const float* w2     = (const float*)d_in[5];   // (E, D, H) fp32
    float*  out = (float*)d_out;                   // (T, D) fp32
    __bf16* hws = (__bf16*)d_ws;                   // (E, TPE, H) bf16 = 8 MiB scratch

    dim3 g1(TPE / BM, H_DIM / BN, N_EXP);          // (8, 8, 8)
    dim3 g2(TPE / BM, D_DIM / BN, N_EXP);          // (8, 16, 8)
    mx_gemm1_silu<<<g1, dim3(256), 0, stream>>>(x, scales, w13, hws);
    mx_gemm2<<<g2, dim3(256), 0, stream>>>(hws, w2, out);

    (void)in_sizes; (void)n_in; (void)out_size; (void)ws_size;
}
